// WeightLoss_80779744903931
// MI455X (gfx1250) — matrix operand order repaired and checked by reading, not compiled, not run on hardware
//
#include <hip/hip_runtime.h>
#include <math.h>

typedef __attribute__((ext_vector_type(2)))  float    v2f;
typedef __attribute__((ext_vector_type(4)))  float    v4f;
typedef __attribute__((ext_vector_type(8)))  float    v8f;
typedef _Float16 half_t;
typedef __attribute__((ext_vector_type(8)))  _Float16 v8h;
typedef __attribute__((ext_vector_type(16))) _Float16 v16h;

#define NROWS 8192
#define NDIM  512
#define NTILES (NROWS / 16)       // 512 16-wide tiles
#define NWAVES 8
#define ALPHA_ 50.0f
#define BETA_  2.0f
#define BASE_  0.5f
#define MARG_  0.1f
#define BIGF   3.0e38f

// f16x3 kernel geometry
#define RB    32                  // rows per block (2 row tiles)
#define NRT   2
#define NBLK  (NROWS / RB)        // 256 blocks
#define LDSH  520                 // padded halves per LDS row (bank-friendly)

// f32 fallback geometry
#define LDSS  516                 // padded floats per LDS row

// ---- wave32 cross-lane reductions ------------------------------------------
__device__ __forceinline__ float grpMin16(float v) {
#pragma unroll
  for (int m = 1; m < 16; m <<= 1) v = fminf(v, __shfl_xor(v, m, 32));
  return v;
}
__device__ __forceinline__ float grpMax16(float v) {
#pragma unroll
  for (int m = 1; m < 16; m <<= 1) v = fmaxf(v, __shfl_xor(v, m, 32));
  return v;
}
__device__ __forceinline__ float grpSum16(float v) {
#pragma unroll
  for (int m = 1; m < 16; m <<= 1) v += __shfl_xor(v, m, 32);
  return v;
}
__device__ __forceinline__ float waveSum32(float v) {
#pragma unroll
  for (int m = 1; m < 32; m <<= 1) v += __shfl_xor(v, m, 32);
  return v;
}

__device__ __forceinline__ v16h cat8(v8h a, v8h b) {
  return __builtin_shufflevector(a, b, 0, 1, 2, 3, 4, 5, 6, 7,
                                 8, 9, 10, 11, 12, 13, 14, 15);
}

// ============================================================================
// Split-fp32 prep: X -> (hi, lo) f16 planes.  x = hi + lo exactly to ~2^-22.
// ============================================================================
__global__ void split_fp32_kernel(const float* __restrict__ X,
                                  half_t* __restrict__ Xhi,
                                  half_t* __restrict__ Xlo) {
  const size_t total = (size_t)NROWS * NDIM;
  for (size_t i = (size_t)blockIdx.x * blockDim.x + threadIdx.x; i < total;
       i += (size_t)gridDim.x * blockDim.x) {
    const float x = X[i];
    const half_t h = (half_t)x;            // RTNE
    Xhi[i] = h;
    Xlo[i] = (half_t)(x - (float)h);       // exact residual, then RTNE
  }
}

// ============================================================================
// Main f16x3 kernel: V_WMMA_F32_16X16X32_F16, 32 rows/block, two passes.
// ============================================================================
__global__ __launch_bounds__(256) void ms_loss_f16x3(
    const half_t* __restrict__ Xhi, const half_t* __restrict__ Xlo,
    const int* __restrict__ T, float* __restrict__ ws) {
  __shared__ half_t sHi[RB * LDSH];        // 33.3 KB
  __shared__ half_t sLo[RB * LDSH];        // 33.3 KB
  __shared__ int    sTrow[RB];
  __shared__ float  sA[NWAVES][RB];
  __shared__ float  sB[NWAVES][RB];
  __shared__ float  sMinPos[RB];
  __shared__ float  sMaxNeg[RB];
  __shared__ float  sLoss[RB];
  __shared__ int    sValid[RB];
  __shared__ float  sStat[NWAVES][4];

  const int tid  = threadIdx.x;
  const int wave = tid >> 5;
  const int lane = tid & 31;
  const int lo   = lane & 15;
  const int hi   = lane >> 4;
  const int rowBase = blockIdx.x * RB;
  const bool lastBlk = (blockIdx.x == NBLK - 1);

  // stage 32x512 hi/lo row block into padded LDS (16B chunks, coalesced)
  for (int i = tid; i < RB * (NDIM / 8); i += 256) {
    const int r = i >> 6;                  // NDIM/8 == 64
    const int c = (i & 63) << 3;
    *(v8h*)&sHi[r * LDSH + c] = *(const v8h*)(Xhi + (size_t)(rowBase + r) * NDIM + c);
    *(v8h*)&sLo[r * LDSH + c] = *(const v8h*)(Xlo + (size_t)(rowBase + r) * NDIM + c);
  }
  if (tid < RB) sTrow[tid] = T[rowBase + tid];
  __syncthreads();

  // A-frag base pointers: lane holds 8-half chunks at kc+8*hi and kc+16+8*hi
  const half_t* aH0 = &sHi[(0 * 16 + lo) * LDSH + 8 * hi];
  const half_t* aL0 = &sLo[(0 * 16 + lo) * LDSH + 8 * hi];
  const half_t* aH1 = &sHi[(1 * 16 + lo) * LDSH + 8 * hi];
  const half_t* aL1 = &sLo[(1 * 16 + lo) * LDSH + 8 * hi];

  // ---------------- pass 1: mining thresholds (+ last-row raw stats) --------
  float minp[NRT][8], maxn[NRT][8];
#pragma unroll
  for (int rt = 0; rt < NRT; ++rt)
#pragma unroll
    for (int r = 0; r < 8; ++r) { minp[rt][r] = BIGF; maxn[rt][r] = -BIGF; }
  float stPS = 0.f, stPC = 0.f, stNS = 0.f, stNC = 0.f;

  for (int ct = wave; ct < NTILES; ct += NWAVES) {
    const int colBase = ct * 16;
    const int tCol = T[colBase + lo];
    const half_t* bH = Xhi + (size_t)(colBase + lo) * NDIM + 8 * hi;
    const half_t* bL = Xlo + (size_t)(colBase + lo) * NDIM + 8 * hi;

    v8f chh0 = {}, chl0 = {}, clh0 = {};   // 6 independent WMMA chains
    v8f chh1 = {}, chl1 = {}, clh1 = {};
#pragma unroll 2
    for (int kc = 0; kc < NDIM; kc += 32) {
      const v16h bh = cat8(*(const v8h*)(bH + kc), *(const v8h*)(bH + kc + 16));
      const v16h bl = cat8(*(const v8h*)(bL + kc), *(const v8h*)(bL + kc + 16));
      const v16h ah0 = cat8(*(const v8h*)(aH0 + kc), *(const v8h*)(aH0 + kc + 16));
      const v16h al0 = cat8(*(const v8h*)(aL0 + kc), *(const v8h*)(aL0 + kc + 16));
      const v16h ah1 = cat8(*(const v8h*)(aH1 + kc), *(const v8h*)(aH1 + kc + 16));
      const v16h al1 = cat8(*(const v8h*)(aL1 + kc), *(const v8h*)(aL1 + kc + 16));
      chh0 = __builtin_amdgcn_wmma_f32_16x16x32_f16(false, ah0, false, bh, (short)0, chh0, false, false);
      chl0 = __builtin_amdgcn_wmma_f32_16x16x32_f16(false, ah0, false, bl, (short)0, chl0, false, false);
      clh0 = __builtin_amdgcn_wmma_f32_16x16x32_f16(false, al0, false, bh, (short)0, clh0, false, false);
      chh1 = __builtin_amdgcn_wmma_f32_16x16x32_f16(false, ah1, false, bh, (short)0, chh1, false, false);
      chl1 = __builtin_amdgcn_wmma_f32_16x16x32_f16(false, ah1, false, bl, (short)0, chl1, false, false);
      clh1 = __builtin_amdgcn_wmma_f32_16x16x32_f16(false, al1, false, bh, (short)0, clh1, false, false);
    }
#pragma unroll
    for (int rt = 0; rt < NRT; ++rt) {
#pragma unroll
      for (int r = 0; r < 8; ++r) {
        const float s = rt ? (chh1[r] + chl1[r] + clh1[r])
                           : (chh0[r] + chl0[r] + clh0[r]);
        const bool same = (sTrow[rt * 16 + r + 8 * hi] == tCol);
        const bool pos  = same && (s < 1.0f);
        const bool neg  = !same;
        minp[rt][r] = pos ? fminf(minp[rt][r], s) : minp[rt][r];
        maxn[rt][r] = neg ? fmaxf(maxn[rt][r], s) : maxn[rt][r];
        if (lastBlk && rt == 1 && r == 7) {   // global row 8191: rt=1,r=7,hi=1
          if (hi == 1) {
            stPS += pos ? s : 0.f;  stPC += pos ? 1.f : 0.f;
            stNS += neg ? s : 0.f;  stNC += neg ? 1.f : 0.f;
          }
        }
      }
    }
  }

#pragma unroll
  for (int rt = 0; rt < NRT; ++rt)
#pragma unroll
    for (int r = 0; r < 8; ++r) {
      const float gm = grpMin16(minp[rt][r]);
      const float gM = grpMax16(maxn[rt][r]);
      if (lo == 0) {
        sA[wave][rt * 16 + r + 8 * hi] = gm;
        sB[wave][rt * 16 + r + 8 * hi] = gM;
      }
    }
  __syncthreads();
  if (tid < RB) {
    float mp = BIGF, mn = -BIGF;
#pragma unroll
    for (int w = 0; w < NWAVES; ++w) {
      mp = fminf(mp, sA[w][tid]);
      mn = fmaxf(mn, sB[w][tid]);
    }
    sMinPos[tid] = mp; sMaxNeg[tid] = mn;
  }
  __syncthreads();

  // ---------------- pass 2: mined exp sums ----------------------------------
  float posS[NRT][8], negS[NRT][8];
#pragma unroll
  for (int rt = 0; rt < NRT; ++rt)
#pragma unroll
    for (int r = 0; r < 8; ++r) { posS[rt][r] = 0.f; negS[rt][r] = 0.f; }

  for (int ct = wave; ct < NTILES; ct += NWAVES) {
    const int colBase = ct * 16;
    const int tCol = T[colBase + lo];
    const half_t* bH = Xhi + (size_t)(colBase + lo) * NDIM + 8 * hi;
    const half_t* bL = Xlo + (size_t)(colBase + lo) * NDIM + 8 * hi;

    v8f chh0 = {}, chl0 = {}, clh0 = {};
    v8f chh1 = {}, chl1 = {}, clh1 = {};
#pragma unroll 2
    for (int kc = 0; kc < NDIM; kc += 32) {
      const v16h bh = cat8(*(const v8h*)(bH + kc), *(const v8h*)(bH + kc + 16));
      const v16h bl = cat8(*(const v8h*)(bL + kc), *(const v8h*)(bL + kc + 16));
      const v16h ah0 = cat8(*(const v8h*)(aH0 + kc), *(const v8h*)(aH0 + kc + 16));
      const v16h al0 = cat8(*(const v8h*)(aL0 + kc), *(const v8h*)(aL0 + kc + 16));
      const v16h ah1 = cat8(*(const v8h*)(aH1 + kc), *(const v8h*)(aH1 + kc + 16));
      const v16h al1 = cat8(*(const v8h*)(aL1 + kc), *(const v8h*)(aL1 + kc + 16));
      chh0 = __builtin_amdgcn_wmma_f32_16x16x32_f16(false, ah0, false, bh, (short)0, chh0, false, false);
      chl0 = __builtin_amdgcn_wmma_f32_16x16x32_f16(false, ah0, false, bl, (short)0, chl0, false, false);
      clh0 = __builtin_amdgcn_wmma_f32_16x16x32_f16(false, al0, false, bh, (short)0, clh0, false, false);
      chh1 = __builtin_amdgcn_wmma_f32_16x16x32_f16(false, ah1, false, bh, (short)0, chh1, false, false);
      chl1 = __builtin_amdgcn_wmma_f32_16x16x32_f16(false, ah1, false, bl, (short)0, chl1, false, false);
      clh1 = __builtin_amdgcn_wmma_f32_16x16x32_f16(false, al1, false, bh, (short)0, clh1, false, false);
    }
#pragma unroll
    for (int rt = 0; rt < NRT; ++rt) {
#pragma unroll
      for (int r = 0; r < 8; ++r) {
        const float s = rt ? (chh1[r] + chl1[r] + clh1[r])
                           : (chh0[r] + chl0[r] + clh0[r]);
        const int  row  = rt * 16 + r + 8 * hi;
        const bool same = (sTrow[row] == tCol);
        const bool pos  = same && (s < 1.0f);
        const bool neg  = !same;
        if (neg && (s + MARG_ > sMinPos[row])) negS[rt][r] += expf(ALPHA_ * (s - BASE_));
        if (pos && (s - MARG_ < sMaxNeg[row])) posS[rt][r] += expf(-BETA_ * (s - BASE_));
      }
    }
  }

#pragma unroll
  for (int rt = 0; rt < NRT; ++rt)
#pragma unroll
    for (int r = 0; r < 8; ++r) {
      const float gp = grpSum16(posS[rt][r]);
      const float gn = grpSum16(negS[rt][r]);
      if (lo == 0) {
        sA[wave][rt * 16 + r + 8 * hi] = gp;
        sB[wave][rt * 16 + r + 8 * hi] = gn;
      }
    }
  if (lastBlk) {
    const float s0 = waveSum32(stPS), s1 = waveSum32(stPC);
    const float s2 = waveSum32(stNS), s3 = waveSum32(stNC);
    if (lane == 0) {
      sStat[wave][0] = s0; sStat[wave][1] = s1;
      sStat[wave][2] = s2; sStat[wave][3] = s3;
    }
  }
  __syncthreads();

  if (tid < RB) {
    float ps = 0.f, ns = 0.f;
#pragma unroll
    for (int w = 0; w < NWAVES; ++w) { ps += sA[w][tid]; ns += sB[w][tid]; }
    const bool hasPos = sMinPos[tid] < (0.5f * BIGF);
    const bool valid  = hasPos && (ps > 0.f) && (ns > 0.f);
    const float rl = (1.0f / BETA_) * log1pf(ps) + (1.0f / ALPHA_) * log1pf(ns);
    sLoss[tid]  = valid ? rl : 0.f;
    sValid[tid] = valid ? 1 : 0;
  }
  __syncthreads();

  if (tid == 0) {
    float ls = 0.f; int vc = 0;
#pragma unroll
    for (int i = 0; i < RB; ++i) { ls += sLoss[i]; vc += sValid[i]; }
    atomicAdd(&ws[0], ls);
    atomicAdd((int*)ws + 1, vc);
    if (lastBlk) {
      float p0 = 0.f, p1 = 0.f, p2 = 0.f, p3 = 0.f;
#pragma unroll
      for (int w = 0; w < NWAVES; ++w) {
        p0 += sStat[w][0]; p1 += sStat[w][1];
        p2 += sStat[w][2]; p3 += sStat[w][3];
      }
      ws[2] = p0; ws[3] = p1; ws[4] = p2; ws[5] = p3;
    }
  }
}

// ============================================================================
// Fallback: pure-fp32 V_WMMA_F32_16X16X4_F32 kernel (no scratch planes needed)
// ============================================================================
__device__ __forceinline__ v8f tile_gemm_f32(const float* __restrict__ aBase,
                                             const float* __restrict__ bBase) {
  v8f c0 = {}, c1 = {};
#pragma unroll 4
  for (int kc = 0; kc < NDIM; kc += 8) {
    v2f a0 = *(const v2f*)(aBase + kc);
    v2f b0 = *(const v2f*)(bBase + kc);
    c0 = __builtin_amdgcn_wmma_f32_16x16x4_f32(false, a0, false, b0, (short)0, c0, false, false);
    v2f a1 = *(const v2f*)(aBase + kc + 4);
    v2f b1 = *(const v2f*)(bBase + kc + 4);
    c1 = __builtin_amdgcn_wmma_f32_16x16x4_f32(false, a1, false, b1, (short)0, c1, false, false);
  }
#pragma unroll
  for (int r = 0; r < 8; ++r) c0[r] += c1[r];
  return c0;
}

__global__ __launch_bounds__(256) void ms_loss_f32(
    const float* __restrict__ X, const int* __restrict__ T,
    float* __restrict__ ws) {
  __shared__ float sX[16 * LDSS];
  __shared__ int   sTrow[16];
  __shared__ float sA[NWAVES][16];
  __shared__ float sB[NWAVES][16];
  __shared__ float sMinPos[16];
  __shared__ float sMaxNeg[16];
  __shared__ float sLoss[16];
  __shared__ int   sValid[16];
  __shared__ float sStat[NWAVES][4];

  const int tid  = threadIdx.x;
  const int wave = tid >> 5;
  const int lane = tid & 31;
  const int lo   = lane & 15;
  const int hi   = lane >> 4;
  const int rowBase = blockIdx.x * 16;
  const bool lastBlk = (blockIdx.x == NTILES - 1);

  for (int i = tid; i < 16 * (NDIM / 4); i += 256) {
    const int r = i >> 7;
    const int c = (i & 127) << 2;
    *(v4f*)&sX[r * LDSS + c] = *(const v4f*)(X + (size_t)(rowBase + r) * NDIM + c);
  }
  if (tid < 16) sTrow[tid] = T[rowBase + tid];
  __syncthreads();

  int tRow[8];
#pragma unroll
  for (int r = 0; r < 8; ++r) tRow[r] = sTrow[r + 8 * hi];
  const float* aBase = &sX[lo * LDSS + 2 * hi];

  float minp[8], maxn[8];
#pragma unroll
  for (int r = 0; r < 8; ++r) { minp[r] = BIGF; maxn[r] = -BIGF; }
  float stPS = 0.f, stPC = 0.f, stNS = 0.f, stNC = 0.f;

  for (int ct = wave; ct < NTILES; ct += NWAVES) {
    const int colBase = ct * 16;
    const int tCol = T[colBase + lo];
    const float* bBase = X + (size_t)(colBase + lo) * NDIM + 2 * hi;
    v8f c = tile_gemm_f32(aBase, bBase);
#pragma unroll
    for (int r = 0; r < 8; ++r) {
      const float s = c[r];
      const bool same = (tRow[r] == tCol);
      const bool pos  = same && (s < 1.0f);
      const bool neg  = !same;
      minp[r] = pos ? fminf(minp[r], s) : minp[r];
      maxn[r] = neg ? fmaxf(maxn[r], s) : maxn[r];
      if (lastBlk && r == 7 && hi == 1) {
        stPS += pos ? s : 0.f;  stPC += pos ? 1.f : 0.f;
        stNS += neg ? s : 0.f;  stNC += neg ? 1.f : 0.f;
      }
    }
  }

#pragma unroll
  for (int r = 0; r < 8; ++r) {
    const float gm = grpMin16(minp[r]);
    const float gM = grpMax16(maxn[r]);
    if (lo == 0) { sA[wave][r + 8 * hi] = gm; sB[wave][r + 8 * hi] = gM; }
  }
  __syncthreads();
  if (tid < 16) {
    float mp = BIGF, mn = -BIGF;
#pragma unroll
    for (int w = 0; w < NWAVES; ++w) {
      mp = fminf(mp, sA[w][tid]);
      mn = fmaxf(mn, sB[w][tid]);
    }
    sMinPos[tid] = mp; sMaxNeg[tid] = mn;
  }
  __syncthreads();

  float posS[8], negS[8];
#pragma unroll
  for (int r = 0; r < 8; ++r) { posS[r] = 0.f; negS[r] = 0.f; }

  for (int ct = wave; ct < NTILES; ct += NWAVES) {
    const int colBase = ct * 16;
    const int tCol = T[colBase + lo];
    const float* bBase = X + (size_t)(colBase + lo) * NDIM + 2 * hi;
    v8f c = tile_gemm_f32(aBase, bBase);
#pragma unroll
    for (int r = 0; r < 8; ++r) {
      const float s = c[r];
      const int  row  = r + 8 * hi;
      const bool same = (tRow[r] == tCol);
      const bool pos  = same && (s < 1.0f);
      const bool neg  = !same;
      if (neg && (s + MARG_ > sMinPos[row])) negS[r] += expf(ALPHA_ * (s - BASE_));
      if (pos && (s - MARG_ < sMaxNeg[row])) posS[r] += expf(-BETA_ * (s - BASE_));
    }
  }

#pragma unroll
  for (int r = 0; r < 8; ++r) {
    const float gp = grpSum16(posS[r]);
    const float gn = grpSum16(negS[r]);
    if (lo == 0) { sA[wave][r + 8 * hi] = gp; sB[wave][r + 8 * hi] = gn; }
  }
  if (lastBlk) {
    const float s0 = waveSum32(stPS), s1 = waveSum32(stPC);
    const float s2 = waveSum32(stNS), s3 = waveSum32(stNC);
    if (lane == 0) {
      sStat[wave][0] = s0; sStat[wave][1] = s1;
      sStat[wave][2] = s2; sStat[wave][3] = s3;
    }
  }
  __syncthreads();

  if (tid < 16) {
    float ps = 0.f, ns = 0.f;
#pragma unroll
    for (int w = 0; w < NWAVES; ++w) { ps += sA[w][tid]; ns += sB[w][tid]; }
    const bool hasPos = sMinPos[tid] < (0.5f * BIGF);
    const bool valid  = hasPos && (ps > 0.f) && (ns > 0.f);
    const float rl = (1.0f / BETA_) * log1pf(ps) + (1.0f / ALPHA_) * log1pf(ns);
    sLoss[tid]  = valid ? rl : 0.f;
    sValid[tid] = valid ? 1 : 0;
  }
  __syncthreads();

  if (tid == 0) {
    float ls = 0.f; int vc = 0;
#pragma unroll
    for (int i = 0; i < 16; ++i) { ls += sLoss[i]; vc += sValid[i]; }
    atomicAdd(&ws[0], ls);
    atomicAdd((int*)ws + 1, vc);
    if (lastBlk) {
      float p0 = 0.f, p1 = 0.f, p2 = 0.f, p3 = 0.f;
#pragma unroll
      for (int w = 0; w < NWAVES; ++w) {
        p0 += sStat[w][0]; p1 += sStat[w][1];
        p2 += sStat[w][2]; p3 += sStat[w][3];
      }
      ws[2] = p0; ws[3] = p1; ws[4] = p2; ws[5] = p3;
    }
  }
}

__global__ void ms_finalize(const float* __restrict__ ws, float* __restrict__ out) {
  const float loss = ws[0] / (float)NROWS;
  const int   vc   = ((const int*)ws)[1];
  const float prec = (float)(NROWS - vc) / (float)NROWS;
  const float pc   = fmaxf(ws[3], 1.0f);
  const float nc   = fmaxf(ws[5], 1.0f);
  out[0] = loss;
  out[1] = prec;
  out[2] = ws[2] / pc;
  out[3] = ws[4] / nc;
}

extern "C" void kernel_launch(void* const* d_in, const int* in_sizes, int n_in,
                              void* d_out, int out_size, void* d_ws, size_t ws_size,
                              hipStream_t stream) {
  const float* X = (const float*)d_in[0];
  const int*   T = (const int*)d_in[1];
  float* ws = (float*)d_ws;
  hipMemsetAsync(d_ws, 0, 64, stream);

  const size_t PLANE = (size_t)NROWS * NDIM;
  const size_t need  = 1024 + 2 * PLANE * sizeof(half_t);   // ~16 MB
  if (ws_size >= need) {
    half_t* Xhi = (half_t*)((char*)d_ws + 1024);
    half_t* Xlo = Xhi + PLANE;
    split_fp32_kernel<<<2048, 256, 0, stream>>>(X, Xhi, Xlo);
    ms_loss_f16x3<<<NBLK, 256, 0, stream>>>(Xhi, Xlo, T, ws);
  } else {
    ms_loss_f32<<<NTILES, 256, 0, stream>>>(X, T, ws);
  }
  ms_finalize<<<1, 1, 0, stream>>>(ws, (float*)d_out);
}
